// LNNElectrodeValueBasedPredictionModel_26671746908681
// MI455X (gfx1250) — matrix operand order repaired and checked by reading, not compiled, not run on hardware
//
#include <hip/hip_runtime.h>

typedef __attribute__((ext_vector_type(16))) _Float16 v16h;
typedef __attribute__((ext_vector_type(8)))  _Float16 v8h;
typedef __attribute__((ext_vector_type(8)))  float    v8f;

#define B_ 256
#define T_ 512
#define D_ 64

// Layers (d_in, h, K=d_in+h, Kpad, Ksteps(K/32), Ntiles(ceil(h/16))):
// L0: 64, 116, 180, 192, 6, 8
// L1: 116, 76, 192, 192, 6, 5
// L2: 76,  64, 140, 160, 5, 4
// LDS weight image: per matrix m in {w1*mask, w2*mask, wa+wb}:
//   element offset = ((m*KS + ks)*NT + nt)*512 + lane*16 + j   (f16 units)
//   holds W[n][k], n = nt*16 + (lane&15), k = ks*32 + ((j>>3)<<4) + ((lane>>4)<<3) + (j&7)
#define W0_OFF 0
#define W1_OFF 73728
#define W2_OFF 119808
#define W_TOTAL 150528

struct Params {
  const float* seq;
  const float* w[3][8];            // w1,w2,wa,wb,b1,b2,ba,bb
  const unsigned char* mask[3];    // bool masks (1 byte/elem)
  float* out;
};

__device__ __forceinline__ float fexp(float x)  { return __builtin_amdgcn_exp2f(x * 1.44269504f); }
__device__ __forceinline__ float frcp_(float x) { return __builtin_amdgcn_rcpf(x); }
__device__ __forceinline__ float fsigmoid(float x) { return frcp_(1.f + fexp(-x)); }
__device__ __forceinline__ float ftanh(float x) {
#if __has_builtin(__builtin_amdgcn_tanhf)
  return __builtin_amdgcn_tanhf(x);
#else
  return 1.f - 2.f * frcp_(1.f + fexp(2.f * x));
#endif
}

// Three GEMMs sharing one A fragment per k-step, then gated combine.
template<int KS, int NT, int KP>
__device__ __forceinline__ void gemm3(const _Float16* xb, const _Float16* wl,
                                      const float* bias, int lane, int wave,
                                      float* hout)
{
  if (wave >= NT) return;                       // wave-uniform: EXEC stays all-1 for WMMA
  v8f a1 = {0.f,0.f,0.f,0.f,0.f,0.f,0.f,0.f};
  v8f a2 = a1, a3 = a1;
  const int m  = lane & 15;
  const int kr = (lane >> 4) << 3;
  const v16h* W = (const v16h*)wl;
#pragma unroll
  for (int ks = 0; ks < KS; ++ks) {
    const _Float16* ap = xb + m * KP + ks * 32 + kr;
    v8h lo = *(const v8h*)ap;                   // K 0..7  (or 8..15)
    v8h hi = *(const v8h*)(ap + 16);            // K 16..23 (or 24..31)
    v16h a;
#pragma unroll
    for (int j = 0; j < 8; ++j) { a[j] = lo[j]; a[j + 8] = hi[j]; }
    v16h b1 = W[((0 * KS + ks) * NT + wave) * 32 + lane];
    v16h b2 = W[((1 * KS + ks) * NT + wave) * 32 + lane];
    v16h b3 = W[((2 * KS + ks) * NT + wave) * 32 + lane];
    a1 = __builtin_amdgcn_wmma_f32_16x16x32_f16(false, a, false, b1, (short)0, a1, false, false);
    a2 = __builtin_amdgcn_wmma_f32_16x16x32_f16(false, a, false, b2, (short)0, a2, false, false);
    a3 = __builtin_amdgcn_wmma_f32_16x16x32_f16(false, a, false, b3, (short)0, a3, false, false);
  }
  const int nl = wave * 16 + (lane & 15);       // bias depends only on N -> per-lane scalar
  const float bv1 = bias[0 * 128 + nl];
  const float bv2 = bias[1 * 128 + nl];
  const float bv3 = bias[2 * 128 + nl];
#pragma unroll
  for (int v = 0; v < 8; ++v) {
    float f1 = ftanh(a1[v] + bv1);
    float f2 = ftanh(a2[v] + bv2);
    float s  = fsigmoid(a3[v] + bv3);
    hout[v] = f1 + s * (f2 - f1);               // ff1*(1-t) + t*ff2
  }
}

template<int NT, int H>
__device__ __forceinline__ void store2(const float* hout,
                                       _Float16* d0, int kp0, int off0,
                                       _Float16* d1, int kp1, int off1,
                                       int lane, int wave)
{
  if (wave >= NT) return;
  int n = wave * 16 + (lane & 15);
  if (n >= H) return;
  int rb = (lane >> 4) << 3;
#pragma unroll
  for (int v = 0; v < 8; ++v) {
    _Float16 hv = (_Float16)hout[v];
    d0[(rb + v) * kp0 + off0 + n] = hv;
    d1[(rb + v) * kp1 + off1 + n] = hv;
  }
}

template<int NT, int H>
__device__ __forceinline__ void store1(const float* hout,
                                       _Float16* d0, int kp0, int off0,
                                       int lane, int wave)
{
  if (wave >= NT) return;
  int n = wave * 16 + (lane & 15);
  if (n >= H) return;
  int rb = (lane >> 4) << 3;
#pragma unroll
  for (int v = 0; v < 8; ++v)
    d0[(rb + v) * kp0 + off0 + n] = (_Float16)hout[v];
}

__global__ __launch_bounds__(256) void cfc_lnn_kernel(Params P)
{
  __shared__ __align__(32) _Float16 sW[W_TOTAL];   // 301,056 B pre-swizzled f16 weights
  __shared__ __align__(16) _Float16 x0[16 * 192];  // [x_t | h0 | pad]
  __shared__ __align__(16) _Float16 x1[16 * 192];  // [h0  | h1]
  __shared__ __align__(16) _Float16 x2[16 * 160];  // [h1  | h2 | pad]
  __shared__ float sB[3][3][128];                  // b1, b2, ba+bb (zero-padded)

  const int tid  = threadIdx.x;
  const int wg   = blockIdx.x;
  const int lane = tid & 31;
  const int wave = tid >> 5;

  // ---- one-time: load, mask, fold, convert and swizzle weights into LDS ----
  const int dins[3]  = {64, 116, 76};
  const int hs[3]    = {116, 76, 64};
  const int KSs[3]   = {6, 6, 5};
  const int NTs[3]   = {8, 5, 4};
  const int wbase[3] = {W0_OFF, W1_OFF, W2_OFF};

  for (int li = 0; li < 3; ++li) {
    const int K = dins[li] + hs[li];
    const int h = hs[li];
    const int KS = KSs[li], NT = NTs[li];
    const float* w1 = P.w[li][0];
    const float* w2 = P.w[li][1];
    const float* wa = P.w[li][2];
    const float* wb = P.w[li][3];
    const unsigned char* mk = P.mask[li];
    const int total = 3 * KS * NT * 512;
    for (int idx = tid; idx < total; idx += 256) {
      int j   = idx & 15;
      int ln  = (idx >> 4) & 31;
      int blk = idx >> 9;
      int nt  = blk % NT;
      int mks = blk / NT;
      int ks  = mks % KS;
      int mm  = mks / KS;
      int n = nt * 16 + (ln & 15);
      int k = ks * 32 + ((j >> 3) << 4) + ((ln >> 4) << 3) + (j & 7);
      float v = 0.f;
      if (n < h && k < K) {
        int o = n * K + k;
        if (mm == 0)      v = mk[o] ? w1[o] : 0.f;
        else if (mm == 1) v = mk[o] ? w2[o] : 0.f;
        else              v = wa[o] + wb[o];
      }
      sW[wbase[li] + idx] = (_Float16)v;
    }
    const float* b1 = P.w[li][4];
    const float* b2 = P.w[li][5];
    const float* ba = P.w[li][6];
    const float* bb = P.w[li][7];
    for (int n = tid; n < 128; n += 256) {
      float v1 = 0.f, v2 = 0.f, v3 = 0.f;
      if (n < h) { v1 = b1[n]; v2 = b2[n]; v3 = ba[n] + bb[n]; }
      sB[li][0][n] = v1; sB[li][1][n] = v2; sB[li][2][n] = v3;
    }
  }
  for (int i = tid; i < 16 * 192; i += 256) { x0[i] = (_Float16)0.f; x1[i] = (_Float16)0.f; }
  for (int i = tid; i < 16 * 160; i += 256) { x2[i] = (_Float16)0.f; }
  __syncthreads();

  // ---- seed x[0] ----
  const int pr = tid >> 4;            // batch row within tile (0..15)
  const int pc = (tid & 15) * 4;      // feature col (float4 granularity)
  const float* seqbase = P.seq + ((size_t)(wg * 16 + pr) * T_) * D_ + pc;
  {
    float4 xf = *(const float4*)seqbase;
    _Float16* d = &x0[pr * 192 + pc];
    d[0] = (_Float16)xf.x; d[1] = (_Float16)xf.y;
    d[2] = (_Float16)xf.z; d[3] = (_Float16)xf.w;
  }
  __syncthreads();

  // ---- recurrence: 512 serial steps, weights stay in LDS ----
  float h0v[8], h1v[8], h2v[8];
  for (int t = 0; t < T_; ++t) {
    float4 xn = make_float4(0.f, 0.f, 0.f, 0.f);
    const bool pre = (t + 1 < T_);
    if (pre) xn = *(const float4*)(seqbase + (size_t)(t + 1) * D_);   // prefetch overlaps L0 gemm

    gemm3<6, 8, 192>(x0, sW + W0_OFF, &sB[0][0][0], lane, wave, h0v);
    __syncthreads();
    if (pre) {
      _Float16* d = &x0[pr * 192 + pc];
      d[0] = (_Float16)xn.x; d[1] = (_Float16)xn.y;
      d[2] = (_Float16)xn.z; d[3] = (_Float16)xn.w;
    }
    store2<8, 116>(h0v, x1, 192, 0, x0, 192, 64, lane, wave);   // -> L1 input + L0 recurrent
    __syncthreads();

    gemm3<6, 5, 192>(x1, sW + W1_OFF, &sB[1][0][0], lane, wave, h1v);
    __syncthreads();
    store2<5, 76>(h1v, x2, 160, 0, x1, 192, 116, lane, wave);   // -> L2 input + L1 recurrent
    __syncthreads();

    gemm3<5, 4, 160>(x2, sW + W2_OFF, &sB[2][0][0], lane, wave, h2v);
    __syncthreads();
    store1<4, 64>(h2v, x2, 160, 76, lane, wave);                // -> L2 recurrent (motor out)
    __syncthreads();
  }

  // ---- outputs: out = h2 [256,64]; h_new = concat(h0,h1,h2) [256,256] ----
  {
    int r = tid >> 4;
    int cb = (tid & 15) * 16;
    int grow = wg * 16 + r;
    float* out = P.out;
#pragma unroll
    for (int j = 0; j < 16; ++j) {
      int c = cb + j;
      float v;
      if (c < 116)      v = (float)x1[r * 192 + c];              // h0
      else if (c < 192) v = (float)x2[r * 160 + (c - 116)];      // h1
      else              v = (float)x2[r * 160 + 76 + (c - 192)]; // h2
      out[16384 + grow * 256 + c] = v;
      if (c >= 192) out[grow * 64 + (c - 192)] = v;              // motor output
    }
  }
}

extern "C" void kernel_launch(void* const* d_in, const int* in_sizes, int n_in,
                              void* d_out, int out_size, void* d_ws, size_t ws_size,
                              hipStream_t stream)
{
  (void)in_sizes; (void)n_in; (void)d_ws; (void)ws_size; (void)out_size;
  Params P;
  P.seq = (const float*)d_in[0];
  for (int li = 0; li < 3; ++li) {
    for (int j = 0; j < 8; ++j) P.w[li][j] = (const float*)d_in[1 + li * 9 + j];
    P.mask[li] = (const unsigned char*)d_in[1 + li * 9 + 8];
  }
  P.out = (float*)d_out;
  cfc_lnn_kernel<<<16, 256, 0, stream>>>(P);
}
